// EncoderRNN_70007966925152
// MI455X (gfx1250) — matrix operand order repaired and checked by reading, not compiled, not run on hardware
//
#include <hip/hip_runtime.h>
#include <hip/hip_bf16.h>
#include <stdint.h>

typedef __attribute__((ext_vector_type(16))) _Float16 v16h;
typedef __attribute__((ext_vector_type(8)))  float    v8f;

#define Bb 32
#define Ss 1024
#define Ee 512
#define Hh 512
#define Ww 128
#define G3 1536   // 3*H

// ---- workspace layout (bytes) ----
// x_f16   [B][S][E]                 : 33,554,432
// wih16   [2][1536][512] f16        :  3,145,728
// whh16   [2][1536][512] f16        :  3,145,728
// gx16    [2][S][B][1536] f16       : 201,326,592
// hbuf    [2 dir][2 pingpong][B][H] :    262,144
// segtab  [B][S] int                :    131,072
static const size_t OFF_X16 = 0;
static const size_t OFF_WIH = OFF_X16 + (size_t)Bb*Ss*Ee*2;
static const size_t OFF_WHH = OFF_WIH + (size_t)2*G3*Ee*2;
static const size_t OFF_GX  = OFF_WHH + (size_t)2*G3*Hh*2;
static const size_t OFF_H   = OFF_GX  + (size_t)2*Ss*Bb*G3*2;
static const size_t OFF_SEG = OFF_H   + (size_t)2*2*Bb*Hh*4;

// monotone order-preserving float->uint key (for atomicMax-based segment max)
__device__ __forceinline__ unsigned fkey(float x) {
    unsigned u = __float_as_uint(x);
    return (u & 0x80000000u) ? ~u : (u | 0x80000000u);
}
__device__ __forceinline__ float fdec(unsigned k) {
    return (k & 0x80000000u) ? __uint_as_float(k & 0x7fffffffu) : __uint_as_float(~k);
}
#define KEY_NEGMAX 0x00800000u  // fkey(-FLT_MAX)

union FragH { v16h h; uint32_t u[8]; _Float16 e[16]; };

// ---- embedding lookup (padding_idx=0) + f32->f16, write pairs ----
__global__ void k_embed(const int* __restrict__ ids, const float* __restrict__ emb,
                        _Float16* __restrict__ x16) {
    size_t tid = (size_t)blockIdx.x * 256 + threadIdx.x; // B*S*(E/2) threads
    int    e2  = (int)(tid & 255);
    size_t bs  = tid >> 8;
    int idx = ids[bs];
    float a = 0.f, c = 0.f;
    if (idx != 0) {
        const float* row = emb + (size_t)idx * Ee + (size_t)e2 * 2;
        a = row[0]; c = row[1];
    }
    x16[tid * 2]     = (_Float16)a;
    x16[tid * 2 + 1] = (_Float16)c;
}

// ---- convert weights to f16: wih16[dir][n][k], whh16[dir][n][k] ----
__global__ void k_cvt(const float* __restrict__ wihf, const float* __restrict__ wihb,
                      const float* __restrict__ whhf, const float* __restrict__ whhb,
                      _Float16* __restrict__ wih16, _Float16* __restrict__ whh16) {
    size_t t = (size_t)blockIdx.x * 256 + threadIdx.x;
    const size_t N = (size_t)G3 * Ee; // 786432 per dir
    if (t < 2 * N) {
        wih16[t] = (_Float16)((t < N) ? wihf[t] : wihb[t - N]);
    } else {
        size_t u = t - 2 * N;
        whh16[u] = (_Float16)((u < N) ? whhf[u] : whhb[u - N]);
    }
}

// ---- precompute segment id per (b,t): searchsorted(layout[b], t, right)-1 ----
__global__ void k_seg(const int* __restrict__ layout, int* __restrict__ seg) {
    int t  = blockIdx.x * 256 + threadIdx.x; // B*S threads
    int b  = t >> 10;
    int tt = t & (Ss - 1);
    const int* lay = layout + b * (Ww + 1);
    int l = 0, r = Ww + 1;
    while (l < r) { int m = (l + r) >> 1; if (lay[m] <= tt) l = m + 1; else r = m; }
    int w = l - 1;
    if (w < 0) w = 0; if (w > Ww - 1) w = Ww - 1;
    seg[t] = w;
}

// ---- init: output keys to -FLT_MAX key, zero both h ping-pong buffers ----
__global__ void k_init(unsigned* __restrict__ outkeys, float* __restrict__ hbuf) {
    size_t t = (size_t)blockIdx.x * 256 + threadIdx.x;
    if (t < (size_t)Bb * Ww * 2 * Hh) outkeys[t] = KEY_NEGMAX;
    if (t < (size_t)2 * 2 * Bb * Hh)  hbuf[t] = 0.f;
}

// ---- big GEMM: gx[dir][s][b][n] = sum_k x[b][s][k] * Wih[dir][n][k]  (f16 WMMA) ----
// one wave -> 16(M) x 64(N) strip;  M index m = s*B + b (time-major gx).
// Two-slab software pipeline: loads for slab k+1 are issued before the WMMAs of
// slab k retire, so VMEM latency overlaps the matrix pipe (partial loadcnt waits).
__global__ void k_gemm_gx(const _Float16* __restrict__ x16,
                          const _Float16* __restrict__ wih16,
                          _Float16* __restrict__ gx16) {
    int wave = (blockIdx.x * 256 + threadIdx.x) >> 5;
    int lane = threadIdx.x & 31;
    int ns   = wave % 24;  int tmp = wave / 24;
    int mt   = tmp % 2048; int dir = tmp / 2048;
    int m0 = mt * 16, n0 = ns * 64;
    int hi = (lane >> 4) & 1;
    v8f z = {0.f,0.f,0.f,0.f,0.f,0.f,0.f,0.f};
    v8f acc[4] = {z, z, z, z};

    int mm = m0 + (lane & 15);
    int b  = mm & 31, s = mm >> 5;
    const uint32_t* arow  = (const uint32_t*)(x16 + ((size_t)b * Ss + s) * Ee);
    const uint32_t* wbase = (const uint32_t*)(wih16 + (size_t)dir * G3 * Ee);
    const uint32_t* brow[4];
#pragma unroll
    for (int nf = 0; nf < 4; ++nf)
        brow[nf] = wbase + (size_t)(n0 + nf * 16 + (lane & 15)) * (Ee / 2);

    FragH A0, A1, B0[4], B1[4];

    auto load_slab = [&](int kk, FragH& A, FragH (&Bf)[4]) {
        int kb = kk * 16; // dwords per 32-K slab
#pragma unroll
        for (int d = 0; d < 8; ++d)
            A.u[d] = arow[kb + (d >> 2) * 8 + hi * 4 + (d & 3)];
#pragma unroll
        for (int nf = 0; nf < 4; ++nf)
#pragma unroll
            for (int j = 0; j < 8; ++j)
                Bf[nf].u[j] = brow[nf][kb + (j >> 2) * 8 + hi * 4 + (j & 3)];
    };
    auto mma4 = [&](FragH& A, FragH (&Bf)[4]) {
#pragma unroll
        for (int nf = 0; nf < 4; ++nf)
            acc[nf] = __builtin_amdgcn_wmma_f32_16x16x32_f16(
                false, A.h, false, Bf[nf].h, (short)0, acc[nf], false, false);
    };

    load_slab(0, A0, B0);
#pragma unroll 1
    for (int kk = 0; kk < 14; kk += 2) {
        load_slab(kk + 1, A1, B1);
        mma4(A0, B0);
        load_slab(kk + 2, A0, B0);
        mma4(A1, B1);
    }
    load_slab(15, A1, B1);
    mma4(A0, B0);
    mma4(A1, B1);

#pragma unroll
    for (int nf = 0; nf < 4; ++nf)
#pragma unroll
        for (int r = 0; r < 8; ++r) {
            int mmw = m0 + r + hi * 8;
            int bb = mmw & 31, ss = mmw >> 5;
            int n = n0 + nf * 16 + (lane & 15);
            gx16[(((size_t)dir * Ss + ss) * Bb + bb) * G3 + n] = (_Float16)acc[nf][r];
        }
}

// ---- one GRU timestep, both directions fused; wave -> 16x16 tile of h_new ----
// 128 waves: dir(2) x mtile(2) x htile(32).  gh = h @ Whh^T via 3 gate WMMA
// chains, same two-slab pipeline as the big GEMM.
__global__ void k_gru_step(int tau,
                           const _Float16* __restrict__ whh16,
                           const _Float16* __restrict__ gx16,
                           float* __restrict__ hbuf,
                           const int* __restrict__ seg,
                           const float* __restrict__ bihf, const float* __restrict__ bhhf,
                           const float* __restrict__ bihb, const float* __restrict__ bhhb,
                           unsigned* __restrict__ outkeys) {
    int wave = (blockIdx.x * 256 + threadIdx.x) >> 5; // 0..127
    int lane = threadIdx.x & 31;
    int dir = wave >> 6, rem = wave & 63;
    int mt = rem >> 5, ht = rem & 31;
    int m0 = mt * 16, n0 = ht * 16;
    int t   = dir ? (Ss - 1 - tau) : tau;
    int cur = tau & 1, nxt = cur ^ 1;
    float* hin  = hbuf + (size_t)(dir * 2 + cur) * Bb * Hh;
    float* hout = hbuf + (size_t)(dir * 2 + nxt) * Bb * Hh;
    int hi = (lane >> 4) & 1;

    v8f z = {0.f,0.f,0.f,0.f,0.f,0.f,0.f,0.f};
    v8f accR = z, accZ = z, accN = z;

    const float*    hrow  = hin + (size_t)(m0 + (lane & 15)) * Hh;
    const uint32_t* wbase = (const uint32_t*)(whh16 + (size_t)dir * G3 * Hh);
    int nB = n0 + (lane & 15);
    const uint32_t* rows[3];
    rows[0] = wbase + (size_t)(0 * Hh + nB) * (Hh / 2);
    rows[1] = wbase + (size_t)(1 * Hh + nB) * (Hh / 2);
    rows[2] = wbase + (size_t)(2 * Hh + nB) * (Hh / 2);

    FragH A0, A1, B0[3], B1[3];

    auto load_slab = [&](int kk, FragH& A, FragH (&Bf)[3]) {
        int kb = kk * 16;
#pragma unroll
        for (int g = 0; g < 3; ++g)
#pragma unroll
            for (int j = 0; j < 8; ++j)
                Bf[g].u[j] = rows[g][kb + (j >> 2) * 8 + hi * 4 + (j & 3)];
#pragma unroll
        for (int i = 0; i < 16; ++i) {
            int K = kk * 32 + (i >> 3) * 16 + hi * 8 + (i & 7);
            A.e[i] = (_Float16)hrow[K];
        }
    };
    auto mma3 = [&](FragH& A, FragH (&Bf)[3]) {
        accR = __builtin_amdgcn_wmma_f32_16x16x32_f16(false, A.h, false, Bf[0].h, (short)0, accR, false, false);
        accZ = __builtin_amdgcn_wmma_f32_16x16x32_f16(false, A.h, false, Bf[1].h, (short)0, accZ, false, false);
        accN = __builtin_amdgcn_wmma_f32_16x16x32_f16(false, A.h, false, Bf[2].h, (short)0, accN, false, false);
    };

    load_slab(0, A0, B0);
#pragma unroll 1
    for (int kk = 0; kk < 14; kk += 2) {
        load_slab(kk + 1, A1, B1);
        mma3(A0, B0);
        load_slab(kk + 2, A0, B0);
        mma3(A1, B1);
    }
    load_slab(15, A1, B1);
    mma3(A0, B0);
    mma3(A1, B1);

    const float* bih = dir ? bihb : bihf;
    const float* bhh = dir ? bhhb : bhhf;
#pragma unroll
    for (int r = 0; r < 8; ++r) {
        int b = m0 + r + hi * 8;
        int j = n0 + (lane & 15);
        const _Float16* gxr = gx16 + (((size_t)dir * Ss + t) * Bb + b) * G3;
        float gr = (float)gxr[j]          + bih[j]          + accR[r] + bhh[j];
        float gz = (float)gxr[Hh + j]     + bih[Hh + j]     + accZ[r] + bhh[Hh + j];
        float rg = 1.f / (1.f + expf(-gr));
        float zg = 1.f / (1.f + expf(-gz));
        float ng = tanhf((float)gxr[2 * Hh + j] + bih[2 * Hh + j] + rg * (accN[r] + bhh[2 * Hh + j]));
        float hold = hin[(size_t)b * Hh + j];
        float hn = (1.f - zg) * ng + zg * hold;
        hout[(size_t)b * Hh + j] = hn;
        int w = seg[b * Ss + t];
        atomicMax(&outkeys[(((size_t)b * Ww + w) * 2 + dir) * Hh + j], fkey(hn));
    }
}

// ---- final hidden: h after last step lives in pingpong buffer 0 (S even) ----
__global__ void k_hidden(const float* __restrict__ hbuf, float* __restrict__ outh) {
    int t = blockIdx.x * 256 + threadIdx.x; // 2*B*H
    int dir = t / (Bb * Hh), rest = t % (Bb * Hh);
    outh[t] = hbuf[(size_t)(dir * 2 + 0) * Bb * Hh + rest];
}

// ---- decode keys -> floats, zero empty segments ----
__global__ void k_finalize(const int* __restrict__ layout, float* __restrict__ out) {
    size_t tid = (size_t)blockIdx.x * 256 + threadIdx.x; // B*W*2H
    size_t bw = tid / (2 * Hh);
    int w = (int)(bw % Ww), b = (int)(bw / Ww);
    const int* lay = layout + b * (Ww + 1);
    int lo = lay[w], hi = lay[w + 1];
    if (lo < 0) lo = 0; if (hi > Ss) hi = Ss;
    unsigned key = ((unsigned*)out)[tid];
    out[tid] = (hi - lo <= 0) ? 0.f : fdec(key);
}

extern "C" void kernel_launch(void* const* d_in, const int* in_sizes, int n_in,
                              void* d_out, int out_size, void* d_ws, size_t ws_size,
                              hipStream_t stream) {
    const int*   ids    = (const int*)  d_in[0];   // [B,S]
    const int*   layout = (const int*)  d_in[3];   // [B,W+1]
    const float* emb    = (const float*)d_in[4];   // [V,E]
    const float* wihf   = (const float*)d_in[5];
    const float* whhf   = (const float*)d_in[6];
    const float* bihf   = (const float*)d_in[7];
    const float* bhhf   = (const float*)d_in[8];
    const float* wihb   = (const float*)d_in[9];
    const float* whhb   = (const float*)d_in[10];
    const float* bihb   = (const float*)d_in[11];
    const float* bhhb   = (const float*)d_in[12];

    char* ws = (char*)d_ws; // needs ~231 MB
    _Float16* x16   = (_Float16*)(ws + OFF_X16);
    _Float16* wih16 = (_Float16*)(ws + OFF_WIH);
    _Float16* whh16 = (_Float16*)(ws + OFF_WHH);
    _Float16* gx16  = (_Float16*)(ws + OFF_GX);
    float*    hbuf  = (float*)   (ws + OFF_H);
    int*      seg   = (int*)     (ws + OFF_SEG);

    float*    out     = (float*)d_out;                       // maxed [B,W,2H]
    unsigned* outkeys = (unsigned*)d_out;
    float*    outh    = out + (size_t)Bb * Ww * 2 * Hh;      // hidden [2,B,H]

    k_embed  <<<32768, 256, 0, stream>>>(ids, emb, x16);
    k_cvt    <<<12288, 256, 0, stream>>>(wihf, wihb, whhf, whhb, wih16, whh16);
    k_seg    <<<  128, 256, 0, stream>>>(layout, seg);
    k_init   <<<16384, 256, 0, stream>>>(outkeys, hbuf);
    k_gemm_gx<<<12288, 256, 0, stream>>>(x16, wih16, gx16);

    for (int tau = 0; tau < Ss; ++tau)
        k_gru_step<<<16, 256, 0, stream>>>(tau, whh16, gx16, hbuf, seg,
                                           bihf, bhhf, bihb, bhhb, outkeys);

    k_hidden  <<<  128, 256, 0, stream>>>(hbuf, outh);
    k_finalize<<<16384, 256, 0, stream>>>(layout, out);
}
